// ResidualVectorQuantizer_42262478192887
// MI455X (gfx1250) — matrix operand order repaired and checked by reading, not compiled, not run on hardware
//
#include <hip/hip_runtime.h>

typedef __attribute__((ext_vector_type(16))) __bf16 v16bf;
typedef __attribute__((ext_vector_type(8)))  float  v8f;

#define B_    32
#define D_    256
#define N_    2048
#define K_    512
#define V_    6
#define ROWS  65536     // B*N
#define DN    524288    // D*N
#define MTILE 128
#define NBLK  512       // ROWS / MTILE
#define CBS   264       // bf16 stride per code row in scb
#define RST   258       // f32 stride per residual row in LDS

// Async global->LDS copy (CDNA5, ASYNCcnt-tracked), GVS addressing mode:
// mem = SADDR(64b SGPR pair) + VADDR(32b byte offset); LDS dest = VDST VGPR.
__device__ __forceinline__ void async_copy_b128(uint32_t lds_byte, uint32_t gbyte,
                                                const float* base)
{
    asm volatile("global_load_async_to_lds_b128 %0, %1, %2"
                 :: "v"(lds_byte), "v"(gbyte), "s"(base)
                 : "memory");
}
__device__ __forceinline__ void wait_async0()
{
    asm volatile("s_wait_asynccnt 0x0" ::: "memory");
}

__global__ __launch_bounds__(256)
void rvq_kernel(const float* __restrict__ inp, const float* __restrict__ emb,
                float* __restrict__ out, float* __restrict__ wsloss)
{
    __shared__ float  res[MTILE * RST];   // 132096 B master f32 residual
    __shared__ float  raw[64 * 256];      //  65536 B raw f32 codebook chunk (dd-major)
    __shared__ __bf16 scb[64 * CBS];      //  33792 B bf16 transposed codebook chunk
    __shared__ float  snorm[64];
    __shared__ int    tok[MTILE];
    __shared__ float  lred[256];

    const int t     = threadIdx.x;
    const int lane  = t & 31;
    const int wave  = t >> 5;
    const int l16   = lane & 15;
    const bool hi   = lane >= 16;
    const int koffA = hi ? 8 : 0;
    const int koffB = hi ? 16 : 0;

    const int g0 = blockIdx.x * MTILE;

    // IO/update mapping: thread owns (rowU, dd = half + 2*i)
    const int rowU = t & 127;
    const int half = t >> 7;
    const int gU   = g0 + rowU;
    const int bU   = gU >> 11;
    const int nU   = gU & 2047;

    // A-matrix row owned by this lane (A layout: lanes 0-15 and 16-31 both M=0..15)
    const int rowA = wave * 16 + l16;

    const uint32_t rawbase = (uint32_t)(uintptr_t)&raw[0];

    // ---- residual := input ----
    #pragma unroll 8
    for (int i = 0; i < 128; ++i) {
        int dd = half + 2 * i;
        res[rowU * RST + dd] = inp[bU * DN + dd * N_ + nU];
    }
    __syncthreads();

    float lacc = 0.0f;
    float* tokout = out + 16777216;

    for (int v = 0; v < V_; ++v) {
        const float* embv = emb + v * (D_ * K_);
        const uint32_t vbyte = (uint32_t)(v * D_ * K_ * 4);

        // bf16 A fragments from LDS residual
        v16bf afr[8];
        #pragma unroll
        for (int f = 0; f < 8; ++f) {
            #pragma unroll
            for (int j = 0; j < 8; ++j) {
                int K0 = 32*f + koffB + 2*j;
                float2 p = *(const float2*)&res[rowA * RST + K0];
                afr[f][2*j]   = (__bf16)p.x;
                afr[f][2*j+1] = (__bf16)p.y;
            }
        }

        float best[8];
        int   bidx[8];
        #pragma unroll
        for (int i = 0; i < 8; ++i) { best[i] = -3.0e38f; bidx[i] = 0; }

        // prologue: async-stage chunk 0 (raw f32, 64 codes x 256 dims)
        {
            const uint32_t ng0b = 0;  // chunk 0
            #pragma unroll 4
            for (int i = 0; i < 16; ++i) {
                int s  = t + 256 * i;        // 4096 x b128 segments
                int dd = s >> 4, q = s & 15;
                async_copy_b128(rawbase + (uint32_t)(dd * 256 + q * 16),
                                vbyte + (uint32_t)(dd * 2048 + ng0b + q * 16), emb);
            }
        }
        wait_async0();
        __syncthreads();

        for (int chunk = 0; chunk < 8; ++chunk) {
            if (t < 64) snorm[t] = 0.0f;
            __syncthreads();
            {   // convert raw f32 -> transposed bf16 scb, accumulate ||e||^2
                const int n  = t & 63;
                const int dg = t >> 6;
                float ps = 0.0f;
                #pragma unroll 8
                for (int i = 0; i < 64; ++i) {
                    int dd = dg * 64 + i;
                    float val = raw[dd * 64 + n];
                    ps += val * val;
                    scb[n * CBS + dd] = (__bf16)val;
                }
                atomicAdd(&snorm[n], ps);
            }
            __syncthreads();

            // async-stage next chunk while this one computes
            if (chunk < 7) {
                const uint32_t ngb = (uint32_t)((chunk + 1) * 64 * 4);
                #pragma unroll 4
                for (int i = 0; i < 16; ++i) {
                    int s  = t + 256 * i;
                    int dd = s >> 4, q = s & 15;
                    async_copy_b128(rawbase + (uint32_t)(dd * 256 + q * 16),
                                    vbyte + (uint32_t)(dd * 2048 + q * 16) + ngb, emb);
                }
            }

            #pragma unroll
            for (int sub = 0; sub < 4; ++sub) {    // 16 codes per WMMA tile
                v8f c;
                #pragma unroll
                for (int i = 0; i < 8; ++i) c[i] = 0.0f;
                const __bf16* bp = &scb[(sub*16 + l16) * CBS + koffB];
                #pragma unroll
                for (int f = 0; f < 8; ++f) {      // K = D = 256 in 8 steps of 32
                    v16bf bfr = *(const v16bf*)(bp + 32*f);
                    c = __builtin_amdgcn_wmma_f32_16x16x32_bf16(
                            false, afr[f], false, bfr, (short)0, c, false, false);
                }
                const float hn2  = 0.5f * snorm[sub*16 + l16];
                const int   code = chunk*64 + sub*16 + l16;
                #pragma unroll
                for (int i = 0; i < 8; ++i) {
                    float s = c[i] - hn2;          // argmax of x.e - ||e||^2/2
                    if (s > best[i]) { best[i] = s; bidx[i] = code; }
                }
            }
            wait_async0();
            __syncthreads();
        }

        // argmax across the 16 lanes of each half-wave (code stripes)
        #pragma unroll
        for (int off = 8; off >= 1; off >>= 1) {
            #pragma unroll
            for (int i = 0; i < 8; ++i) {
                float os = __shfl_xor(best[i], off, 32);
                int   oi = __shfl_xor(bidx[i], off, 32);
                if (os > best[i] || (os == best[i] && oi < bidx[i])) {
                    best[i] = os; bidx[i] = oi;
                }
            }
        }
        if (l16 == 0) {   // C layout: lane 0 -> rows 0..7, lane 16 -> rows 8..15
            #pragma unroll
            for (int i = 0; i < 8; ++i) {
                int row = (hi ? 8 : 0) + i;
                tok[wave * 16 + row] = bidx[i];
                tokout[v * ROWS + g0 + wave * 16 + row] = (float)bidx[i];
            }
        }
        __syncthreads();

        // residual -= emb[:, idx]; loss += new_residual^2
        const int idxu = tok[rowU];
        #pragma unroll 8
        for (int i = 0; i < 128; ++i) {
            int dd = half + 2 * i;
            float q  = embv[dd * K_ + idxu];
            float nr = res[rowU * RST + dd] - q;
            res[rowU * RST + dd] = nr;
            lacc += nr * nr;
        }
        __syncthreads();
    }

    // q_sum = input - final residual
    #pragma unroll 8
    for (int i = 0; i < 128; ++i) {
        int dd = half + 2 * i;
        int o = bU * DN + dd * N_ + nU;
        out[o] = inp[o] - res[rowU * RST + dd];
    }

    // per-block loss partial
    lred[t] = lacc;
    __syncthreads();
    for (int s = 128; s > 0; s >>= 1) {
        if (t < s) lred[t] += lred[t + s];
        __syncthreads();
    }
    if (t == 0) wsloss[blockIdx.x] = lred[0];
}

__global__ __launch_bounds__(512)
void rvq_loss_reduce(const float* __restrict__ wsloss, float* __restrict__ out)
{
    __shared__ float s[512];
    int t = threadIdx.x;
    s[t] = wsloss[t];
    __syncthreads();
    for (int k = 256; k > 0; k >>= 1) {
        if (t < k) s[t] += s[t + k];
        __syncthreads();
    }
    if (t == 0) out[16777216 + 393216] = s[0] * (1.0f / 33554432.0f);
}

extern "C" void kernel_launch(void* const* d_in, const int* in_sizes, int n_in,
                              void* d_out, int out_size, void* d_ws, size_t ws_size,
                              hipStream_t stream)
{
    const float* inp = (const float*)d_in[0];   // (B, D, N) f32
    const float* emb = (const float*)d_in[1];   // (V, D, K) f32
    float* out = (float*)d_out;                 // q_sum | tokens | loss
    float* ws  = (float*)d_ws;                  // NBLK loss partials

    rvq_kernel<<<NBLK, 256, 0, stream>>>(inp, emb, out, ws);
    rvq_loss_reduce<<<1, NBLK, 0, stream>>>(ws, out);
}
